// MLPAttention_63677185131059
// MI455X (gfx1250) — matrix operand order repaired and checked by reading, not compiled, not run on hardware
//
#include <hip/hip_runtime.h>
#include <hip/hip_bf16.h>

// ---------------------------------------------------------------------------
// MLP attention, fused for MI455X (gfx1250, wave32, WMMA f16 16x16x32).
//
// B=4, TQ=TK=256, Q=K=H=256.  Layer-2 GEMM dominates: 34.4 GFLOP -> f16 WMMA.
// One workgroup per (b,q): build the full 256x256 f16 "h" tile in LDS and
// stage W2^T (f16) in LDS once per block.  Each wave owns TWO 16-row k-tiles
// (A fragments register-resident) so every LDS B fragment feeds two WMMAs.
// relu(+b2)*w3 score reduction is fused into the WMMA epilogue; the
// [B,Tq,Tk,H] intermediate and h2 are never materialized.
// LDS: 2 x 132KB (W2T first = low DS-immediate offsets; both 264-half padded
// stride for conflict-free ds_load_b128) + ~5KB vectors = 269KB / 320KB WGP.
// ---------------------------------------------------------------------------

typedef __attribute__((ext_vector_type(16))) _Float16 v16h;
typedef __attribute__((ext_vector_type(8)))  float    v8f;

#define Bn  4
#define TQ  256
#define TK  256
#define HD  256              // H = Q = K feature dims
#define HT_STRIDE 264        // f16 row stride (16B aligned, 4-bank skew per row)

// ---------------------------------------------------------------------------
// Prep A: kh[b,k,:] = keys[b,k,:] @ W1[:256]    (blocks 0..1023)
//         qh[b,q,:] = queries[b,q,:] @ W1[256:] (blocks 1024..2047)
// ---------------------------------------------------------------------------
__global__ __launch_bounds__(256)
void mlpattn_proj_kernel(const float* __restrict__ queries,
                         const float* __restrict__ keys,
                         const float* __restrict__ W1,
                         float* __restrict__ qh,
                         float* __restrict__ kh) {
    __shared__ float row[HD];
    const int bid = blockIdx.x;
    const int tid = threadIdx.x;

    const float* src;
    const float* w;
    float* dst;
    if (bid < Bn * TK) {                 // kh rows
        src = keys + (size_t)bid * HD;
        w   = W1;                        // W1[:K]
        dst = kh + (size_t)bid * HD;
    } else {                             // qh rows
        const int r = bid - Bn * TK;
        src = queries + (size_t)r * HD;
        w   = W1 + (size_t)HD * HD;      // W1[K:]
        dst = qh + (size_t)r * HD;
    }
    row[tid] = src[tid];
    __syncthreads();

    float acc = 0.f;
#pragma unroll 8
    for (int d = 0; d < HD; ++d)
        acc = fmaf(row[d], w[(size_t)d * HD + tid], acc);
    dst[tid] = acc;
}

// ---------------------------------------------------------------------------
// Prep B: W2T[g][h] = (f16) W2[h][g]  so B fragments are K-contiguous.
// ---------------------------------------------------------------------------
__global__ __launch_bounds__(256)
void mlpattn_w2t_kernel(const float* __restrict__ W2, _Float16* __restrict__ W2T) {
    const int g = blockIdx.x;
    const int h = threadIdx.x;
    W2T[(size_t)g * HD + h] = (_Float16)W2[(size_t)h * HD + g];
}

// ---------------------------------------------------------------------------
// Main fused kernel: one workgroup per (b, q). 256 threads = 8 waves.
// Wave w owns k-rows [w*32, w*32+32) as two 16-row WMMA M-tiles.
// ---------------------------------------------------------------------------
__global__ __launch_bounds__(256)
void mlpattn_main_kernel(const float* __restrict__ keys,
                         const int*   __restrict__ qlen,
                         const int*   __restrict__ klen,
                         const float* __restrict__ b1,
                         const float* __restrict__ b2,
                         const float* __restrict__ W3,
                         const float* __restrict__ qh,
                         const float* __restrict__ kh,
                         const _Float16* __restrict__ W2T,
                         float* __restrict__ out) {
    // W2T first: B-fragment DS addresses stay in the low 132KB, maximizing
    // 16-bit DS-immediate folding in the hot loop.
    __shared__ _Float16 sh_w2t[HD * HT_STRIDE];     // 135168 B (padded W2T copy)
    __shared__ _Float16 sh_h[TK * HT_STRIDE];       // 135168 B
    __shared__ float sh_scores[TK];
    __shared__ float sh_qh[HD];
    __shared__ float sh_b1[HD];
    __shared__ float sh_b2[HD];
    __shared__ float sh_w3[HD];
    __shared__ float sh_red[16];

    const int tid  = threadIdx.x;
    const int lane = tid & 31;
    const int wave = tid >> 5;
    const int bid  = blockIdx.x;
    const int b    = bid >> 8;          // batch
    const int q    = bid & 255;         // query row

    // stage per-(b,q) vectors
    sh_qh[tid] = qh[((size_t)(b * TQ + q)) * HD + tid];
    sh_b1[tid] = b1[tid];
    sh_b2[tid] = b2[tid];
    sh_w3[tid] = W3[tid];               // W3 is [H,1]

    // ---- stage W2T into LDS (once per block, padded stride) ----------------
    {
        const float4* src = reinterpret_cast<const float4*>(W2T);
#pragma unroll 8
        for (int f = tid; f < HD * 32; f += 256) {
            const int r = f >> 5;            // W2T row (g)
            const int c = f & 31;            // float4 within row
            *reinterpret_cast<float4*>(&sh_w2t[r * HT_STRIDE + c * 8]) = src[f];
        }
    }
    __syncthreads();   // sh_qh/sh_b1 ready for the h-tile build

    // ---- build f16 h tile: h[k,:] = relu(qh + kh[b,k,:] + b1) --------------
    // 4-wide: each thread owns 4 consecutive h-columns; float4 global loads
    // (64 threads cover a row) + one packed 8B f16x4 LDS store per row.
    {
        union H4 { float2 f2; _Float16 h[4]; };
        const int h0 = (tid & 63) * 4;
        const int r0 = tid >> 6;             // 4 row streams
        const float4 qv  = *reinterpret_cast<const float4*>(&sh_qh[h0]);
        const float4 b1v = *reinterpret_cast<const float4*>(&sh_b1[h0]);
        const float* khbase = kh + ((size_t)(b * TK)) * HD;
#pragma unroll 4
        for (int kr = r0; kr < TK; kr += 4) {
            const float4 kv =
                *reinterpret_cast<const float4*>(khbase + (size_t)kr * HD + h0);
            H4 p;
            p.h[0] = (_Float16)fmaxf(qv.x + kv.x + b1v.x, 0.f);
            p.h[1] = (_Float16)fmaxf(qv.y + kv.y + b1v.y, 0.f);
            p.h[2] = (_Float16)fmaxf(qv.z + kv.z + b1v.z, 0.f);
            p.h[3] = (_Float16)fmaxf(qv.w + kv.w + b1v.w, 0.f);
            *reinterpret_cast<float2*>(&sh_h[kr * HT_STRIDE + h0]) = p.f2;
        }
    }
    __syncthreads();

    // ---- WMMA: scores[k] = sum_g relu( (h @ W2)[k,g] + b2[g] ) * w3[g] -----
    const int n    = lane & 15;          // N column within 16x16 tile
    const int hsel = (lane >> 4) & 1;    // lane half (K-group select)
    const int m    = lane & 15;          // A row owned by this lane
    const int kr0  = wave * 32;          // first k-row of this wave's pair

    // Preload A fragments for both M-tiles into registers (2 x 8 x 8 VGPRs).
    // 16x32 f16 A layout: lanes 0-15 hold K=0..7,16..23 ; lanes 16-31 K=8..15,24..31
    v16h afr0[8], afr1[8];
    {
        const _Float16* arow0 = &sh_h[(kr0 + m) * HT_STRIDE];
        const _Float16* arow1 = arow0 + 16 * HT_STRIDE;
#pragma unroll
        for (int hc = 0; hc < 8; ++hc) {
            union { v16h v; float4 f[2]; } a0, a1;
            const _Float16* ap0 = arow0 + hc * 32 + hsel * 8;
            const _Float16* ap1 = arow1 + hc * 32 + hsel * 8;
            a0.f[0] = *reinterpret_cast<const float4*>(ap0);
            a0.f[1] = *reinterpret_cast<const float4*>(ap0 + 16);
            a1.f[0] = *reinterpret_cast<const float4*>(ap1);
            a1.f[1] = *reinterpret_cast<const float4*>(ap1 + 16);
            afr0[hc] = a0.v;
            afr1[hc] = a1.v;
        }
    }

    float sacc0[8], sacc1[8];
#pragma unroll
    for (int i = 0; i < 8; ++i) { sacc0[i] = 0.f; sacc1[i] = 0.f; }

#pragma unroll 2
    for (int gt = 0; gt < 16; ++gt) {
        v8f acc0 = {};
        v8f acc1 = {};
        const _Float16* brow = &sh_w2t[(gt * 16 + n) * HT_STRIDE + hsel * 8];
#pragma unroll
        for (int hc = 0; hc < 8; ++hc) {
            // B fragment (32x16 f16): lane n holds column g, K=8h..8h+7, 16+8h..16+8h+7
            union { v16h v; float4 f[2]; } bfr;
            const _Float16* bp = brow + hc * 32;
            bfr.f[0] = *reinterpret_cast<const float4*>(bp);
            bfr.f[1] = *reinterpret_cast<const float4*>(bp + 16);
            acc0 = __builtin_amdgcn_wmma_f32_16x16x32_f16(
                false, afr0[hc], false, bfr.v, (short)0, acc0, false, false);
            acc1 = __builtin_amdgcn_wmma_f32_16x16x32_f16(
                false, afr1[hc], false, bfr.v, (short)0, acc1, false, false);
        }
        // epilogue: relu(+b2) * w3, accumulated per owned row
        const float b2v = sh_b2[gt * 16 + n];
        const float w3v = sh_w3[gt * 16 + n];
#pragma unroll
        for (int i = 0; i < 8; ++i) {
            float v0 = fmaxf(acc0[i] + b2v, 0.f);
            float v1 = fmaxf(acc1[i] + b2v, 0.f);
            sacc0[i] = fmaf(v0, w3v, sacc0[i]);
            sacc1[i] = fmaf(v1, w3v, sacc1[i]);
        }
    }

    // reduce across the 16 lanes sharing each row half
#pragma unroll
    for (int i = 0; i < 8; ++i) {
        float v0 = sacc0[i];
        float v1 = sacc1[i];
        v0 += __shfl_xor(v0, 1, 32);
        v1 += __shfl_xor(v1, 1, 32);
        v0 += __shfl_xor(v0, 2, 32);
        v1 += __shfl_xor(v1, 2, 32);
        v0 += __shfl_xor(v0, 4, 32);
        v1 += __shfl_xor(v1, 4, 32);
        v0 += __shfl_xor(v0, 8, 32);
        v1 += __shfl_xor(v1, 8, 32);
        sacc0[i] = v0;
        sacc1[i] = v1;
    }
    if ((lane & 15) == 0) {
        // C/D layout: lanes 0-15 hold M=0..7 (vgpr i -> M=i), lanes 16-31 M=8..15
        const int mb0 = kr0 + (hsel << 3);
        const int mb1 = kr0 + 16 + (hsel << 3);
#pragma unroll
        for (int i = 0; i < 8; ++i) {
            sh_scores[mb0 + i] = sacc0[i];
            sh_scores[mb1 + i] = sacc1[i];
        }
    }
    __syncthreads();

    // ---- masked softmax over k ---------------------------------------------
    const int kl = klen[b];
    const int ql = qlen[b];
    const float s = (tid < kl) ? sh_scores[tid] : -3.0e38f;

    float mx = s;
    mx = fmaxf(mx, __shfl_xor(mx, 16, 32));
    mx = fmaxf(mx, __shfl_xor(mx, 8, 32));
    mx = fmaxf(mx, __shfl_xor(mx, 4, 32));
    mx = fmaxf(mx, __shfl_xor(mx, 2, 32));
    mx = fmaxf(mx, __shfl_xor(mx, 1, 32));
    if (lane == 0) sh_red[wave] = mx;
    __syncthreads();
    float bm = sh_red[0];
#pragma unroll
    for (int i = 1; i < 8; ++i) bm = fmaxf(bm, sh_red[i]);

    float e = (tid < kl) ? __expf(s - bm) : 0.f;
    float sm = e;
    sm += __shfl_xor(sm, 16, 32);
    sm += __shfl_xor(sm, 8, 32);
    sm += __shfl_xor(sm, 4, 32);
    sm += __shfl_xor(sm, 2, 32);
    sm += __shfl_xor(sm, 1, 32);
    if (lane == 0) sh_red[8 + wave] = sm;
    __syncthreads();
    float tot = 0.f;
#pragma unroll
    for (int i = 0; i < 8; ++i) tot += sh_red[8 + i];

    float prob = (q < ql) ? (e / tot) : 0.f;   // fully-masked query row -> zeros
    sh_scores[tid] = prob;                     // reuse as probs
    __syncthreads();

    // ---- out[b,q,:] = probs @ keys[b] --------------------------------------
    float oacc = 0.f;
    const float* kp = keys + ((size_t)b * TK) * HD + tid;
#pragma unroll 4
    for (int k = 0; k < TK; ++k)
        oacc = fmaf(sh_scores[k], kp[(size_t)k * HD], oacc);
    out[((size_t)(b * TQ + q)) * HD + tid] = oacc;
}

// ---------------------------------------------------------------------------
// Launch
// ---------------------------------------------------------------------------
extern "C" void kernel_launch(void* const* d_in, const int* in_sizes, int n_in,
                              void* d_out, int out_size, void* d_ws, size_t ws_size,
                              hipStream_t stream) {
    const float* queries = (const float*)d_in[0];
    const float* keys    = (const float*)d_in[1];
    const int*   qlen    = (const int*)d_in[2];
    const int*   klen    = (const int*)d_in[3];
    const float* W1      = (const float*)d_in[4];
    const float* b1      = (const float*)d_in[5];
    const float* W2      = (const float*)d_in[6];
    const float* b2      = (const float*)d_in[7];
    const float* W3      = (const float*)d_in[8];
    // b3 (d_in[9]) cancels in softmax; unused.

    float* qh = (float*)d_ws;                              // 1 MB
    float* kh = qh + (size_t)Bn * TQ * HD;                 // 1 MB
    _Float16* W2T = (_Float16*)(kh + (size_t)Bn * TK * HD);// 128 KB

    mlpattn_proj_kernel<<<2 * Bn * TK, 256, 0, stream>>>(queries, keys, W1, qh, kh);
    mlpattn_w2t_kernel<<<HD, 256, 0, stream>>>(W2, W2T);
    mlpattn_main_kernel<<<Bn * TQ, 256, 0, stream>>>(keys, qlen, klen, b1, b2, W3,
                                                     qh, kh, W2T, (float*)d_out);
}
